// SwinTransformerBlock_3822520893899
// MI455X (gfx1250) — matrix operand order repaired and checked by reading, not compiled, not run on hardware
//
#include <hip/hip_runtime.h>
#include <math.h>

// ---------------------------------------------------------------------------
// Types / WMMA helpers (CDNA5 gfx1250, wave32)
// ---------------------------------------------------------------------------
typedef __attribute__((ext_vector_type(16))) __bf16 bf16x16;
typedef __attribute__((ext_vector_type(8)))  float  floatx8;

union BFrag {
  bf16x16 v;
  __bf16  e[16];
  uint4   u[2];
};

static __device__ __forceinline__ floatx8 wmma_bf16(const BFrag a, const BFrag b, floatx8 c) {
  // D(16x16 f32) = A(16x32 bf16) * B(32x16 bf16) + C
  return __builtin_amdgcn_wmma_f32_16x16x32_bf16(false, a.v, false, b.v, (short)0, c, false, false);
}

static __device__ __forceinline__ floatx8 zero8() {
  floatx8 z;
#pragma unroll
  for (int i = 0; i < 8; ++i) z[i] = 0.0f;
  return z;
}

// window token -> flat token index. B=16, H=W=64, WS=8, nW=64, N=64
static __device__ __forceinline__ int tok_index(int b, int w, int i) {
  int wy = w >> 3, wx = w & 7, iy = i >> 3, ix = i & 7;
  return (b << 12) + (((wy << 3) + iy) << 6) + (wx << 3) + ix;
}

static __device__ __forceinline__ unsigned short f2bu(float f) {
  __bf16 b = (__bf16)f;
  return __builtin_bit_cast(unsigned short, b);
}

// ---------------------------------------------------------------------------
// Weight convert + transpose: out[n*K + k] = bf16(in[k*N + n])
// ---------------------------------------------------------------------------
__global__ void wconv_kernel(const float* __restrict__ in, __bf16* __restrict__ out,
                             int K, int N) {
  int tid = blockIdx.x * blockDim.x + threadIdx.x;
  if (tid >= K * N) return;
  int n = tid / K, k = tid % K;
  out[tid] = (__bf16)in[k * N + n];
}

// ---------------------------------------------------------------------------
// LayerNorm over C=256, one token per wave (8 tokens / 256-thread block)
// ---------------------------------------------------------------------------
__global__ __launch_bounds__(256) void ln_kernel(const float* __restrict__ x,
                                                 const float* __restrict__ g,
                                                 const float* __restrict__ bta,
                                                 __bf16* __restrict__ y) {
  int lane = threadIdx.x & 31;
  int t = blockIdx.x * 8 + (threadIdx.x >> 5);
  const float* xp = x + (size_t)t * 256;
  float v[8]; float s = 0.f;
#pragma unroll
  for (int i = 0; i < 8; ++i) { v[i] = xp[lane + 32 * i]; s += v[i]; }
#pragma unroll
  for (int m = 16; m >= 1; m >>= 1) s += __shfl_xor(s, m);
  float mu = s * (1.0f / 256.0f);
  float vs = 0.f;
#pragma unroll
  for (int i = 0; i < 8; ++i) { float d = v[i] - mu; vs += d * d; }
#pragma unroll
  for (int m = 16; m >= 1; m >>= 1) vs += __shfl_xor(vs, m);
  float rs = rsqrtf(vs * (1.0f / 256.0f) + 1e-5f);
  __bf16* yp = y + (size_t)t * 256;
#pragma unroll
  for (int i = 0; i < 8; ++i) {
    int c = lane + 32 * i;
    yp[c] = (__bf16)((v[i] - mu) * rs * g[c] + bta[c]);
  }
}

// ---------------------------------------------------------------------------
// Window pooling -> prototypes (block = one (b,w) window, thread = channel)
// ---------------------------------------------------------------------------
__global__ __launch_bounds__(256) void pool_kernel(const __bf16* __restrict__ qn,
                                                   const __bf16* __restrict__ sn,
                                                   const float* __restrict__ mask,
                                                   float* __restrict__ q_prot,
                                                   float* __restrict__ s_prot,
                                                   float* __restrict__ msum) {
  int bw = blockIdx.x; int b = bw >> 6, w = bw & 63;
  int c = threadIdx.x;
  float qs = 0.f, ss = 0.f, ms = 0.f;
  for (int i = 0; i < 64; ++i) {
    int t = tok_index(b, w, i);
    float m = mask[t];
    qs += (float)qn[(size_t)t * 256 + c];
    ss += (float)sn[(size_t)t * 256 + c] * m;
    ms += m;
  }
  q_prot[(size_t)bw * 256 + c] = qs * (1.0f / 64.0f);
  s_prot[(size_t)bw * 256 + c] = ss / (ms + 5e-4f);
  if (c == 0) msum[bw] = ms;
}

// ---------------------------------------------------------------------------
// Cosine matching + argmax over support windows (block = batch, thread = qw)
// ---------------------------------------------------------------------------
__global__ void match_kernel(const float* __restrict__ q_prot,
                             const float* __restrict__ s_prot,
                             const float* __restrict__ msum,
                             int* __restrict__ qs_index) {
  int b = blockIdx.x, qw = threadIdx.x;
  const float* qp = q_prot + (size_t)(b * 64 + qw) * 256;
  float qn2 = 0.f;
  for (int c = 0; c < 256; ++c) qn2 += qp[c] * qp[c];
  float qn_ = sqrtf(qn2);
  float best = -1e30f; int bi = 0;
  for (int sw = 0; sw < 64; ++sw) {
    const float* sp = s_prot + (size_t)(b * 64 + sw) * 256;
    float dot = 0.f, sn2 = 0.f;
    for (int c = 0; c < 256; ++c) { float sv = sp[c]; dot += sv * qp[c]; sn2 += sv * sv; }
    float cs = dot / (sqrtf(sn2) * qn_ + 1e-7f);
    cs = (cs + 1.0f) * 0.5f * (msum[b * 64 + sw] > 0.0f ? 1.0f : 0.0f);
    if (cs > best) { best = cs; bi = sw; }
  }
  qs_index[b * 64 + qw] = bi;
}

// ---------------------------------------------------------------------------
// bf16 WMMA GEMM:  out = A[MxK] @ W[KxN] + bias (+res / gelu)
//   A row-major bf16; Bt = W transposed to [N x K] bf16 (B-fragment friendly)
//   epi: 0 = bias -> f32 ; 1 = bias + residual -> f32 ; 2 = bias + gelu -> bf16
// Each wave computes a 32x32 tile (2x2 WMMA fragments), 8 waves per block.
// ---------------------------------------------------------------------------
__global__ __launch_bounds__(256) void gemm_bf16_kernel(const __bf16* __restrict__ A,
                                                        const __bf16* __restrict__ Bt,
                                                        const float* __restrict__ bias,
                                                        const float* __restrict__ res,
                                                        float* __restrict__ outF,
                                                        __bf16* __restrict__ outB,
                                                        int M, int N, int K, int epi) {
  const int lane = threadIdx.x & 31;
  const int lh = lane >> 4, l15 = lane & 15;
  const int tilesN = N >> 5;
  int wt = blockIdx.x * 8 + (threadIdx.x >> 5);
  int tm = (wt / tilesN) << 5;
  int tn = (wt % tilesN) << 5;
  (void)M;

  floatx8 c00 = zero8(), c01 = zero8(), c10 = zero8(), c11 = zero8();

  for (int kk = 0; kk < K; kk += 32) {
    BFrag a0, a1, b0, b1;
    {
      const __bf16* p = A + (size_t)(tm + l15) * K + kk;
      a0.u[0] = *reinterpret_cast<const uint4*>(p + 16 * lh);
      a0.u[1] = *reinterpret_cast<const uint4*>(p + 8 + 16 * lh);
    }
    {
      const __bf16* p = A + (size_t)(tm + 16 + l15) * K + kk;
      a1.u[0] = *reinterpret_cast<const uint4*>(p + 16 * lh);
      a1.u[1] = *reinterpret_cast<const uint4*>(p + 8 + 16 * lh);
    }
    {
      const __bf16* p = Bt + (size_t)(tn + l15) * K + kk + 16 * lh;
      b0.u[0] = *reinterpret_cast<const uint4*>(p);
      b0.u[1] = *reinterpret_cast<const uint4*>(p + 8);
    }
    {
      const __bf16* p = Bt + (size_t)(tn + 16 + l15) * K + kk + 16 * lh;
      b1.u[0] = *reinterpret_cast<const uint4*>(p);
      b1.u[1] = *reinterpret_cast<const uint4*>(p + 8);
    }
    c00 = wmma_bf16(a0, b0, c00);
    c01 = wmma_bf16(a0, b1, c01);
    c10 = wmma_bf16(a1, b0, c10);
    c11 = wmma_bf16(a1, b1, c11);
  }

  floatx8 acc[2][2] = {{c00, c01}, {c10, c11}};
#pragma unroll
  for (int i = 0; i < 2; ++i)
#pragma unroll
    for (int j = 0; j < 2; ++j) {
      int n = tn + j * 16 + l15;
      float bs = bias[n];
#pragma unroll
      for (int r = 0; r < 8; ++r) {
        int m = tm + i * 16 + (lh << 3) + r;
        size_t idx = (size_t)m * N + n;
        float v = acc[i][j][r] + bs;
        if (epi == 1) v += res[idx];
        if (epi == 2) {
          v = 0.5f * v * (1.0f + erff(v * 0.7071067811865476f));  // exact GELU
          outB[idx] = (__bf16)v;
        } else {
          outF[idx] = v;
        }
      }
    }
}

// ---------------------------------------------------------------------------
// Joint self+cross window attention.
// Block = one (batch, window); wave = one head. ctx = [win w of KV0 (self,
// scaled), win w2 of KV1 (cross, cosine-normalized)]. Joint softmax, P@V.
// Q: f32 [B*L, 256] (cols = h*32+d); KV*: f32 [B*L, 512] (K cols 0..255, V 256..511)
// ---------------------------------------------------------------------------
__global__ __launch_bounds__(256) void attn_kernel(const float* __restrict__ Q,
                                                   const float* __restrict__ KV0,
                                                   const float* __restrict__ KV1,
                                                   const int* __restrict__ sel,
                                                   __bf16* __restrict__ outB) {
  __shared__ float qn_s[8][64];
  __shared__ float kn_s[8][64];
  __shared__ __align__(16) unsigned short p_s[8][2048];  // per-wave 16x128 bf16 P

  const int b = blockIdx.x >> 6;
  const int w = blockIdx.x & 63;
  const int w2 = sel ? sel[(b << 6) + w] : w;
  const int h = threadIdx.x >> 5;
  const int lane = threadIdx.x & 31;
  const int lh = lane >> 4, l15 = lane & 15;

  // --- per-row Q norms (64 rows) and per-ctx-col K norms (cross half) ---
#pragma unroll
  for (int u = 0; u < 2; ++u) {
    int rr = lane * 2 + u;
    const float* qp = Q + (size_t)tok_index(b, w, rr) * 256 + h * 32;
    float s = 0.f;
#pragma unroll
    for (int d = 0; d < 32; ++d) s += qp[d] * qp[d];
    qn_s[h][rr] = sqrtf(s);
    const float* kp = KV1 + (size_t)tok_index(b, w2, rr) * 512 + h * 32;
    s = 0.f;
#pragma unroll
    for (int d = 0; d < 32; ++d) s += kp[d] * kp[d];
    kn_s[h][rr] = sqrtf(s);
  }

  // --- K B-fragments: 8 ctx tiles of 16 tokens, K-dim = head_dim 32 ---
  BFrag kf[8];
#pragma unroll
  for (int t = 0; t < 8; ++t) {
    int ci = t * 16 + l15;
    const float* kp = (ci < 64)
        ? KV0 + (size_t)tok_index(b, w, ci) * 512 + h * 32 + lh * 16
        : KV1 + (size_t)tok_index(b, w2, ci - 64) * 512 + h * 32 + lh * 16;
#pragma unroll
    for (int j = 0; j < 16; ++j) kf[t].e[j] = (__bf16)kp[j];
  }

  // --- V B-fragments: rows = ctx (128 = 4 ksteps), cols = head dim (2 tiles) ---
  BFrag vf[4][2];
#pragma unroll
  for (int kk = 0; kk < 4; ++kk)
#pragma unroll
    for (int j2 = 0; j2 < 2; ++j2) {
      int d = j2 * 16 + l15;
#pragma unroll
      for (int hh = 0; hh < 16; ++hh) {
        int ci = kk * 32 + lh * 16 + hh;
        const float* vp = (ci < 64)
            ? KV0 + (size_t)tok_index(b, w, ci) * 512 + 256 + h * 32
            : KV1 + (size_t)tok_index(b, w2, ci - 64) * 512 + 256 + h * 32;
        vf[kk][j2].e[hh] = (__bf16)vp[d];
      }
    }

  // --- 4 strips of 16 query rows ---
  for (int mi = 0; mi < 4; ++mi) {
    BFrag aq;
    {
      const float* qp = Q + (size_t)tok_index(b, w, mi * 16 + l15) * 256 + h * 32;
#pragma unroll
      for (int j = 0; j < 8; ++j) {
        aq.e[j]     = (__bf16)qp[16 * lh + j];
        aq.e[8 + j] = (__bf16)qp[8 + 16 * lh + j];
      }
    }

    floatx8 sc[8];
#pragma unroll
    for (int t = 0; t < 8; ++t) sc[t] = wmma_bf16(aq, kf[t], zero8());

    float qnr[8];
#pragma unroll
    for (int r = 0; r < 8; ++r) qnr[r] = qn_s[h][mi * 16 + (lh << 3) + r];

#pragma unroll
    for (int t = 0; t < 4; ++t)
#pragma unroll
      for (int r = 0; r < 8; ++r) sc[t][r] *= 0.17677669529663687f;  // 32^-0.5
#pragma unroll
    for (int t = 4; t < 8; ++t) {
      float kn = kn_s[h][(t - 4) * 16 + l15];
#pragma unroll
      for (int r = 0; r < 8; ++r) sc[t][r] = sc[t][r] / (qnr[r] * kn + 1e-7f);
    }

    // row-wise softmax over 128 (lane halves hold independent row sets)
    float mx[8], sm[8];
#pragma unroll
    for (int r = 0; r < 8; ++r) mx[r] = -3.4e38f;
#pragma unroll
    for (int t = 0; t < 8; ++t)
#pragma unroll
      for (int r = 0; r < 8; ++r) mx[r] = fmaxf(mx[r], sc[t][r]);
#pragma unroll
    for (int r = 0; r < 8; ++r) {
      mx[r] = fmaxf(mx[r], __shfl_xor(mx[r], 1));
      mx[r] = fmaxf(mx[r], __shfl_xor(mx[r], 2));
      mx[r] = fmaxf(mx[r], __shfl_xor(mx[r], 4));
      mx[r] = fmaxf(mx[r], __shfl_xor(mx[r], 8));
      sm[r] = 0.f;
    }
#pragma unroll
    for (int t = 0; t < 8; ++t)
#pragma unroll
      for (int r = 0; r < 8; ++r) {
        float e = __expf(sc[t][r] - mx[r]);
        sc[t][r] = e;
        sm[r] += e;
      }
#pragma unroll
    for (int r = 0; r < 8; ++r) {
      sm[r] += __shfl_xor(sm[r], 1);
      sm[r] += __shfl_xor(sm[r], 2);
      sm[r] += __shfl_xor(sm[r], 4);
      sm[r] += __shfl_xor(sm[r], 8);
      sm[r] = 1.0f / sm[r];
    }

    // write P (C-layout) -> LDS bf16, row-major 16x128
#pragma unroll
    for (int t = 0; t < 8; ++t)
#pragma unroll
      for (int r = 0; r < 8; ++r) {
        int ml = (lh << 3) + r;
        p_s[h][ml * 128 + t * 16 + l15] = f2bu(sc[t][r] * sm[r]);
      }

    // P @ V : read back in A-fragment layout
    floatx8 o0 = zero8(), o1 = zero8();
#pragma unroll
    for (int kk = 0; kk < 4; ++kk) {
      BFrag pa;
      const unsigned short* pr = &p_s[h][l15 * 128 + kk * 32 + 16 * lh];
      pa.u[0] = *reinterpret_cast<const uint4*>(pr);
      pa.u[1] = *reinterpret_cast<const uint4*>(pr + 8);
      o0 = wmma_bf16(pa, vf[kk][0], o0);
      o1 = wmma_bf16(pa, vf[kk][1], o1);
    }

    // store attention output (bf16, cols = h*32 + d)
#pragma unroll
    for (int r = 0; r < 8; ++r) {
      int row = mi * 16 + (lh << 3) + r;
      size_t base = (size_t)tok_index(b, w, row) * 256 + h * 32;
      outB[base + l15]      = (__bf16)o0[r];
      outB[base + 16 + l15] = (__bf16)o1[r];
    }
  }
}

// ---------------------------------------------------------------------------
// Host orchestration
// ---------------------------------------------------------------------------
extern "C" void kernel_launch(void* const* d_in, const int* in_sizes, int n_in,
                              void* d_out, int out_size, void* d_ws, size_t ws_size,
                              hipStream_t stream) {
  (void)in_sizes; (void)n_in; (void)out_size; (void)ws_size;

  const size_t TOK = 65536;            // B * H * W
  const size_t BLC = TOK * 256;        // elements per (B,L,C) tensor

  const float* q    = (const float*)d_in[0];
  const float* s    = (const float*)d_in[1];
  const float* mask = (const float*)d_in[2];
  const float* n1g  = (const float*)d_in[5];
  const float* n1b  = (const float*)d_in[6];
  const float* n2g  = (const float*)d_in[7];
  const float* n2b  = (const float*)d_in[8];
  const float* Wq_q  = (const float*)d_in[9];
  const float* bq_q  = (const float*)d_in[10];
  const float* Wkv_q = (const float*)d_in[11];
  const float* bkv_q = (const float*)d_in[12];
  const float* Wp_q  = (const float*)d_in[13];
  const float* bp_q  = (const float*)d_in[14];
  const float* Wq_s  = (const float*)d_in[15];
  const float* bq_s  = (const float*)d_in[16];
  const float* Wkv_s = (const float*)d_in[17];
  const float* bkv_s = (const float*)d_in[18];
  const float* Wp_s  = (const float*)d_in[19];
  const float* bp_s  = (const float*)d_in[20];
  const float* W1_q  = (const float*)d_in[21];
  const float* b1_q  = (const float*)d_in[22];
  const float* W2_q  = (const float*)d_in[23];
  const float* b2_q  = (const float*)d_in[24];
  const float* W1_s  = (const float*)d_in[25];
  const float* b1_s  = (const float*)d_in[26];
  const float* W2_s  = (const float*)d_in[27];
  const float* b2_s  = (const float*)d_in[28];

  float* outq = (float*)d_out;
  float* outs = outq + BLC;

  // ---- workspace carve ----
  char* cur = (char*)d_ws;
  auto carve = [&](size_t bytes) -> void* {
    void* r = (void*)cur;
    cur += (bytes + 255) & ~(size_t)255;
    return r;
  };
  __bf16* qn_bf  = (__bf16*)carve(BLC * 2);
  __bf16* sn_bf  = (__bf16*)carve(BLC * 2);
  float*  Qbuf   = (float*)carve(BLC * 4);
  float*  KVA    = (float*)carve(TOK * 512 * 4);
  float*  KVB    = (float*)carve(TOK * 512 * 4);
  __bf16* attnbf = (__bf16*)carve(BLC * 2);
  __bf16* x1nbf  = (__bf16*)carve(BLC * 2);
  __bf16* Hbuf   = (__bf16*)carve(TOK * 1024 * 2);
  float*  q_prot = (float*)carve(1024 * 256 * 4);
  float*  s_prot = (float*)carve(1024 * 256 * 4);
  float*  msum   = (float*)carve(1024 * 4);
  int*    qsidx  = (int*)carve(1024 * 4);
  __bf16* WqT_q  = (__bf16*)carve(65536 * 2);
  __bf16* WkvT_q = (__bf16*)carve(131072 * 2);
  __bf16* WpT_q  = (__bf16*)carve(65536 * 2);
  __bf16* WqT_s  = (__bf16*)carve(65536 * 2);
  __bf16* WkvT_s = (__bf16*)carve(131072 * 2);
  __bf16* WpT_s  = (__bf16*)carve(65536 * 2);
  __bf16* W1T_q  = (__bf16*)carve(262144 * 2);
  __bf16* W2T_q  = (__bf16*)carve(262144 * 2);
  __bf16* W1T_s  = (__bf16*)carve(262144 * 2);
  __bf16* W2T_s  = (__bf16*)carve(262144 * 2);

  auto wconv = [&](const float* in, __bf16* out, int K, int N) {
    int n = K * N;
    wconv_kernel<<<(n + 255) / 256, 256, 0, stream>>>(in, out, K, N);
  };
  auto gemm = [&](const __bf16* A, const __bf16* Bt, const float* bias,
                  const float* res, float* oF, __bf16* oB, int M, int N, int K, int epi) {
    int blocks = (M / 32) * (N / 32) / 8;
    gemm_bf16_kernel<<<blocks, 256, 0, stream>>>(A, Bt, bias, res, oF, oB, M, N, K, epi);
  };

  // 1) weights -> bf16, transposed to [N x K]
  wconv(Wq_q, WqT_q, 256, 256);   wconv(Wkv_q, WkvT_q, 256, 512);
  wconv(Wp_q, WpT_q, 256, 256);   wconv(Wq_s, WqT_s, 256, 256);
  wconv(Wkv_s, WkvT_s, 256, 512); wconv(Wp_s, WpT_s, 256, 256);
  wconv(W1_q, W1T_q, 256, 1024);  wconv(W2_q, W2T_q, 1024, 256);
  wconv(W1_s, W1T_s, 256, 1024);  wconv(W2_s, W2T_s, 1024, 256);

  // 2) norm1
  ln_kernel<<<TOK / 8, 256, 0, stream>>>(q, n1g, n1b, qn_bf);
  ln_kernel<<<TOK / 8, 256, 0, stream>>>(s, n1g, n1b, sn_bf);

  // 3) prototypes + matching
  pool_kernel<<<1024, 256, 0, stream>>>(qn_bf, sn_bf, mask, q_prot, s_prot, msum);
  match_kernel<<<16, 64, 0, stream>>>(q_prot, s_prot, msum, qsidx);

  // ---- q branch: ctx = [q_w, s_sel_w] ----
  gemm(qn_bf, WqT_q,  bq_q,  nullptr, Qbuf, nullptr, 65536, 256, 256, 0);
  gemm(qn_bf, WkvT_q, bkv_q, nullptr, KVA,  nullptr, 65536, 512, 256, 0);
  gemm(sn_bf, WkvT_q, bkv_q, nullptr, KVB,  nullptr, 65536, 512, 256, 0);
  attn_kernel<<<1024, 256, 0, stream>>>(Qbuf, KVA, KVB, qsidx, attnbf);
  gemm(attnbf, WpT_q, bp_q, q, outq, nullptr, 65536, 256, 256, 1);  // + shortcut_q

  // ---- s branch: ctx = [s_w, q_w] ----
  gemm(sn_bf, WqT_s,  bq_s,  nullptr, Qbuf, nullptr, 65536, 256, 256, 0);
  gemm(sn_bf, WkvT_s, bkv_s, nullptr, KVA,  nullptr, 65536, 512, 256, 0);
  gemm(qn_bf, WkvT_s, bkv_s, nullptr, KVB,  nullptr, 65536, 512, 256, 0);
  attn_kernel<<<1024, 256, 0, stream>>>(Qbuf, KVA, KVB, nullptr, attnbf);
  gemm(attnbf, WpT_s, bp_s, s, outs, nullptr, 65536, 256, 256, 1);  // + shortcut_s

  // ---- q MLP: out += W2( gelu( W1( LN2(out) ) ) ) ----
  ln_kernel<<<TOK / 8, 256, 0, stream>>>(outq, n2g, n2b, x1nbf);
  gemm(x1nbf, W1T_q, b1_q, nullptr, nullptr, Hbuf, 65536, 1024, 256, 2);
  gemm(Hbuf,  W2T_q, b2_q, outq, outq, nullptr, 65536, 256, 1024, 1);

  // ---- s MLP ----
  ln_kernel<<<TOK / 8, 256, 0, stream>>>(outs, n2g, n2b, x1nbf);
  gemm(x1nbf, W1T_s, b1_s, nullptr, nullptr, Hbuf, 65536, 1024, 256, 2);
  gemm(Hbuf,  W2T_s, b2_s, outs, outs, nullptr, 65536, 256, 1024, 1);
}
